// NTXentLossWithMemoryBank_12824772346325
// MI455X (gfx1250) — matrix operand order repaired and checked by reading, not compiled, not run on hardware
//
#include <hip/hip_runtime.h>
#include <cmath>

typedef __attribute__((ext_vector_type(16))) __bf16 v16bf;
typedef __attribute__((ext_vector_type(8)))  float  v8f;

#define B_ROWS 1024
#define DIM    256
#define K_BANK 65536
#define NCHUNK 65          // 65 * 1024 = 66560 = B_ROWS + K_BANK
#define CHUNK  1024
#define NTILES (CHUNK / 16)
#define MTILE  128         // rows per block = 8 waves * 16

#define SCALEF 2.8853900817779268f  // log2(e)/T, T = 0.5
#define LN2F   0.6931471805599453f

// ---- workspace layout (bytes) ----
#define WS_Z2N  0                            // 1024*256*4      = 1048576
#define WS_AHI  (WS_Z2N + B_ROWS*DIM*4)      // 1024*256*2      =  524288
#define WS_ALO  (WS_AHI + B_ROWS*DIM*2)      // 1024*256*2      =  524288
#define WS_POS  (WS_ALO + B_ROWS*DIM*2)      // 1024*4          =    4096
#define WS_MAX  (WS_POS + B_ROWS*4)          // scalar (padded to 256)
#define WS_PART (WS_MAX + 256)               // 1024*65*4       =  266240

static __device__ inline float wave_sum32(float v) {
#pragma unroll
  for (int m = 16; m >= 1; m >>= 1) v += __shfl_xor(v, m, 32);
  return v;
}

// zero the atomic-max scalar (ws is poisoned with 0xAA and never re-poisoned)
__global__ void ntx_init_kernel(unsigned int* maxbits) { *maxbits = 0u; }

// One wave per row: L2-normalize z1/z2, emit z2n (f32), z1n hi/lo (bf16), pos logit.
__global__ void ntx_prep_kernel(const float* __restrict__ z1,
                                const float* __restrict__ z2,
                                float* __restrict__ z2n,
                                __bf16* __restrict__ ahi,
                                __bf16* __restrict__ alo,
                                float* __restrict__ pos2) {
  const int wid  = threadIdx.x >> 5;
  const int lane = threadIdx.x & 31;
  const int row  = blockIdx.x * 8 + wid;

  float x1[8], x2[8];
  float s1 = 0.f, s2 = 0.f;
#pragma unroll
  for (int j = 0; j < 8; ++j) {
    const int e = lane + 32 * j;
    x1[j] = z1[(size_t)row * DIM + e];
    x2[j] = z2[(size_t)row * DIM + e];
    s1 += x1[j] * x1[j];
    s2 += x2[j] * x2[j];
  }
  s1 = wave_sum32(s1);
  s2 = wave_sum32(s2);
  const float i1 = 1.f / fmaxf(sqrtf(s1), 1e-12f);
  const float i2 = 1.f / fmaxf(sqrtf(s2), 1e-12f);

  float dp = 0.f;
#pragma unroll
  for (int j = 0; j < 8; ++j) {
    const int e = lane + 32 * j;
    const float a = x1[j] * i1;
    const float b = x2[j] * i2;
    dp += a * b;
    z2n[(size_t)row * DIM + e] = b;
    const __bf16 h = (__bf16)a;
    ahi[(size_t)row * DIM + e] = h;
    alo[(size_t)row * DIM + e] = (__bf16)(a - (float)h);
  }
  dp = wave_sum32(dp);
  if (lane == 0) pos2[row] = dp * SCALEF;  // base-2 scaled logit
}

// One wave per bank row: max of ||bank_j||^2 via deterministic uint atomicMax.
__global__ void ntx_bankmax_kernel(const float* __restrict__ bank,
                                   unsigned int* __restrict__ maxbits) {
  const int wid  = threadIdx.x >> 5;
  const int lane = threadIdx.x & 31;
  const int row  = blockIdx.x * 8 + wid;
  float ss = 0.f;
#pragma unroll
  for (int j = 0; j < 8; ++j) {
    const float v = bank[(size_t)row * DIM + lane + 32 * j];
    ss += v * v;
  }
  ss = wave_sum32(ss);
  if (lane == 0) atomicMax(maxbits, __float_as_uint(ss));  // ss >= 0
}

// Main: bf16x3 WMMA GEMM, double-buffered LDS B tile, software-pipelined loads,
// fixed-shift exp accumulation.
// grid = (NCHUNK, 8), block = 256 (8 waves; wave w owns rows mbase..mbase+15).
__global__ __launch_bounds__(256) void ntx_main_kernel(
    const __bf16* __restrict__ ahi, const __bf16* __restrict__ alo,
    const float* __restrict__ z2n, const float* __restrict__ bank,
    const float* __restrict__ maxss, float* __restrict__ part) {
  __shared__ __bf16 Bhi[2][16][264];  // +8 bf16 pad to stagger LDS banks
  __shared__ __bf16 Blo[2][16][264];

  const int wid  = threadIdx.x >> 5;
  const int lane = threadIdx.x & 31;
  const int h    = lane >> 4;   // wave half
  const int l16  = lane & 15;
  const int mbase = blockIdx.y * MTILE + wid * 16;
  const int arow  = mbase + l16;           // A: both halves hold same row
  const int cbase = blockIdx.x * CHUNK;    // first global column of this block

  const float mss = *maxss;
  const float M2  = fmaxf(sqrtf(mss), 1.0f) * SCALEF;  // global logit bound (base-2)

  // A fragments resident in registers for the whole column sweep.
  v16bf Ah[8], Al[8];
  {
    const __bf16* ph = ahi + (size_t)arow * DIM;
    const __bf16* pl = alo + (size_t)arow * DIM;
#pragma unroll
    for (int ks = 0; ks < 8; ++ks) {
      const int k0 = 32 * ks;
#pragma unroll
      for (int e = 0; e < 8; ++e) {
        Ah[ks][e]     = ph[k0 + 8 * h + e];
        Ah[ks][8 + e] = ph[k0 + 16 + 8 * h + e];
        Al[ks][e]     = pl[k0 + 8 * h + e];
        Al[ks][8 + e] = pl[k0 + 16 + 8 * h + e];
      }
    }
  }

  float sacc[8];
#pragma unroll
  for (int r = 0; r < 8; ++r) sacc[r] = 0.f;

  // B-tile loader mapping: 256 threads -> 16 rows x 16 element-chunks
  const int brow = threadIdx.x >> 4;
  const int boff = (threadIdx.x & 15) * 16;

  // prologue: tile 0 -> buffer 0
  {
    const int gj = cbase + brow;
    const float* wr = (gj < B_ROWS) ? (z2n + (size_t)gj * DIM)
                                    : (bank + (size_t)(gj - B_ROWS) * DIM);
#pragma unroll
    for (int i = 0; i < 16; ++i) {
      const float v = wr[boff + i];
      const __bf16 hv = (__bf16)v;
      Bhi[0][brow][boff + i] = hv;
      Blo[0][brow][boff + i] = (__bf16)(v - (float)hv);
    }
  }
  __syncthreads();

  for (int t = 0; t < NTILES; ++t) {
    const int p  = t & 1;
    const int jb = cbase + t * 16;  // global column of tile row 0

    // Issue next tile's global loads before compute so they drain behind WMMAs.
    float stage[16];
    if (t + 1 < NTILES) {
      const int gj = jb + 16 + brow;
      const float* wr = (gj < B_ROWS) ? (z2n + (size_t)gj * DIM)
                                      : (bank + (size_t)(gj - B_ROWS) * DIM);
#pragma unroll
      for (int i = 0; i < 16; ++i) stage[i] = wr[boff + i];
    }

    // Compute from buffer p.
    v8f c = {};
#pragma unroll
    for (int ks = 0; ks < 8; ++ks) {
      const int k0 = 32 * ks;
      v16bf bh, bl;
#pragma unroll
      for (int e = 0; e < 8; ++e) {
        bh[e]     = Bhi[p][l16][k0 + 8 * h + e];
        bh[8 + e] = Bhi[p][l16][k0 + 16 + 8 * h + e];
        bl[e]     = Blo[p][l16][k0 + 8 * h + e];
        bl[8 + e] = Blo[p][l16][k0 + 16 + 8 * h + e];
      }
      c = __builtin_amdgcn_wmma_f32_16x16x32_bf16(false, Ah[ks], false, bh,
                                                  (short)0, c, false, false);
      c = __builtin_amdgcn_wmma_f32_16x16x32_bf16(false, Ah[ks], false, bl,
                                                  (short)0, c, false, false);
      c = __builtin_amdgcn_wmma_f32_16x16x32_bf16(false, Al[ks], false, bh,
                                                  (short)0, c, false, false);
    }

#pragma unroll
    for (int r = 0; r < 8; ++r) {
      const int grow = mbase + 8 * h + r;        // C row for this lane/vgpr
      const bool msk = (blockIdx.x == 0) && ((jb + l16) == grow);  // diag of inbatch
      const float x = c[r] * SCALEF - M2;
      sacc[r] += msk ? 0.f : exp2f(x);
    }

    // Convert staged tile into the alternate buffer.
    if (t + 1 < NTILES) {
      const int q = p ^ 1;
#pragma unroll
      for (int i = 0; i < 16; ++i) {
        const __bf16 hv = (__bf16)stage[i];
        Bhi[q][brow][boff + i] = hv;
        Blo[q][brow][boff + i] = (__bf16)(stage[i] - (float)hv);
      }
    }
    __syncthreads();
  }

  // Butterfly sum across the 16 lanes of each half (columns), then store partials.
#pragma unroll
  for (int r = 0; r < 8; ++r) {
#pragma unroll
    for (int m = 8; m >= 1; m >>= 1) sacc[r] += __shfl_xor(sacc[r], m, 32);
  }
  if (l16 == 0) {
#pragma unroll
    for (int r = 0; r < 8; ++r)
      part[(size_t)(mbase + 8 * h + r) * NCHUNK + blockIdx.x] = sacc[r];
  }
}

// Single-block deterministic final reduction: fold partials + positive term.
__global__ void ntx_reduce_kernel(const float* __restrict__ part,
                                  const float* __restrict__ pos2,
                                  const float* __restrict__ maxss,
                                  float* __restrict__ out) {
  __shared__ float red[256];
  const float M2 = fmaxf(sqrtf(*maxss), 1.0f) * SCALEF;
  float acc = 0.f;
  for (int row = threadIdx.x; row < B_ROWS; row += 256) {
    float s = exp2f(pos2[row] - M2);  // logits[:,0] term
    for (int cchunk = 0; cchunk < NCHUNK; ++cchunk)
      s += part[(size_t)row * NCHUNK + cchunk];
    acc += LN2F * (M2 + log2f(s) - pos2[row]);  // logsumexp_e - pos_e
  }
  red[threadIdx.x] = acc;
  __syncthreads();
#pragma unroll
  for (int stride = 128; stride >= 1; stride >>= 1) {
    if (threadIdx.x < stride) red[threadIdx.x] += red[threadIdx.x + stride];
    __syncthreads();
  }
  if (threadIdx.x == 0) out[0] = red[0] * (1.0f / B_ROWS);
}

extern "C" void kernel_launch(void* const* d_in, const int* in_sizes, int n_in,
                              void* d_out, int out_size, void* d_ws, size_t ws_size,
                              hipStream_t stream) {
  (void)in_sizes; (void)n_in; (void)out_size; (void)ws_size;
  const float* z1   = (const float*)d_in[0];
  const float* z2   = (const float*)d_in[1];
  const float* bank = (const float*)d_in[2];
  float* out = (float*)d_out;

  char* ws = (char*)d_ws;
  float*        z2n     = (float*)(ws + WS_Z2N);
  __bf16*       ahi     = (__bf16*)(ws + WS_AHI);
  __bf16*       alo     = (__bf16*)(ws + WS_ALO);
  float*        pos2    = (float*)(ws + WS_POS);
  unsigned int* maxbits = (unsigned int*)(ws + WS_MAX);
  float*        maxss   = (float*)(ws + WS_MAX);
  float*        part    = (float*)(ws + WS_PART);

  ntx_init_kernel<<<1, 1, 0, stream>>>(maxbits);
  ntx_prep_kernel<<<B_ROWS / 8, 256, 0, stream>>>(z1, z2, z2n, ahi, alo, pos2);
  ntx_bankmax_kernel<<<K_BANK / 8, 256, 0, stream>>>(bank, maxbits);
  ntx_main_kernel<<<dim3(NCHUNK, B_ROWS / MTILE), 256, 0, stream>>>(
      ahi, alo, z2n, bank, maxss, part);
  ntx_reduce_kernel<<<1, 256, 0, stream>>>(part, pos2, maxss, out);
}
